// AttnDecoder_4569845203516
// MI455X (gfx1250) — matrix operand order repaired and checked by reading, not compiled, not run on hardware
//
#include <hip/hip_runtime.h>
#include <hip/hip_bf16.h>

// ---------------------------------------------------------------------------
// Problem dimensions (match reference)
// ---------------------------------------------------------------------------
#define TT  64
#define BB  64
#define SS  256
#define EE  512
#define HH  1024
#define HEE 1024
#define XK  (EE + HEE + HH)   // 2560 : [emb | cont | h] concat K for gates GEMM
#define GN  (4 * HH)          // 4096 : gate width

// WMMA GEMM tiling: 8 waves (wave32) per block, 2(M) x 4(N) wave grid,
// each wave computes a 16x64 C tile (four v_wmma_f32_16x16x32_bf16 per K-step)
#define BM 32
#define BN 256
#define KC 32
#define LDP (KC + 8)          // LDS row pad for A (40*2 = 80 B rows, 16B aligned)

typedef __attribute__((ext_vector_type(16))) __bf16 v16bf;
typedef __attribute__((ext_vector_type(8)))  float  v8f;
typedef __attribute__((ext_vector_type(4)))  unsigned int v4u;
typedef __attribute__((ext_vector_type(8)))  unsigned int v8u;

// ---------------------------------------------------------------------------
// Generic GEMM:  C[m*ldc+n] = sum_k A[m,k] * W[n,k]  (+ bias[n])
//   A: f32, converted to bf16 while staging to LDS (VALU path).
//   W: pre-packed bf16, n-major / k-contiguous; B tile staged to LDS by the
//      Tensor Data Mover (wave 0 issues tensor_load_to_lds, TENSORcnt sync).
//   A row address = (m % aRowsInner)*aStrideInner + (m / aRowsInner)*aStrideOuter
// ---------------------------------------------------------------------------
__global__ __launch_bounds__(256)
void k_gemm_bf16(float* __restrict__ C, int ldc,
                 const float* __restrict__ A,
                 int aRowsInner, long long aStrideInner, long long aStrideOuter,
                 const __bf16* __restrict__ Bp,
                 const float* __restrict__ bias,
                 int M, int N, int K)
{
    __shared__ __bf16 As[BM][LDP];
    __shared__ __bf16 Bs[BN][KC];   // dense: TDM writes a contiguous tile

    const int tid  = threadIdx.x;
    const int lane = tid & 31;
    const int wave = tid >> 5;
    const int wm   = wave & 1;       // wave M index 0..1
    const int wn   = wave >> 1;      // wave N index 0..3 (64 cols each)
    const int half = lane >> 4;      // lane group (0: lanes 0-15, 1: 16-31)
    const int l    = lane & 15;

    const int mBlock = blockIdx.y * BM;
    const int nBlock = blockIdx.x * BN;

    // A stager: thread -> 4 consecutive f32 of one row
    const int arow = tid >> 3;              // 0..31
    const int acol = (tid & 7) << 2;        // 0,4,..,28
    const int mG   = mBlock + arow;
    const size_t aBase = (size_t)(mG % aRowsInner) * (size_t)aStrideInner
                       + (size_t)(mG / aRowsInner) * (size_t)aStrideOuter;

    // LDS byte offset of the B tile (for the TDM descriptor)
    const unsigned ldsB = (unsigned)(size_t)(void*)&Bs[0][0];

    v8f acc[4] = {v8f{}, v8f{}, v8f{}, v8f{}};

    for (int k0 = 0; k0 < K; k0 += KC) {
        // ---- B tile: Tensor Data Mover, issued once per block by wave 0 ----
        // 2-D descriptor: elements = bf16 (data_size=1 -> 2B), tile KCxBN,
        // dim0 stride = K elements. Group0: count=1, lds_addr, global_addr,
        // type=2.  Group1: data_size / dims / strides per ISA 8.4.
        if (wave == 0) {
            unsigned long long ga =
                (unsigned long long)(size_t)(Bp + (size_t)nBlock * K + k0);
            v4u g0;
            g0[0] = 1u;                                   // count=1, user mode
            g0[1] = ldsB;                                 // lds_addr
            g0[2] = (unsigned)ga;                         // global_addr[31:0]
            g0[3] = (unsigned)((ga >> 32) & 0x01ffffffull)// global_addr[56:32]
                  | (2u << 30);                           // type=2 (image)
            v8u g1;
            g1[0] = 0x00010000u;                          // data_size=1 (2B)
            g1[1] = (unsigned)KC << 16;                   // tensor_dim0 = KC
            g1[2] = (unsigned)BN << 16;                   // tensor_dim1 = BN
            g1[3] = (unsigned)KC << 16;                   // tile_dim0 = KC
            g1[4] = (unsigned)BN;                         // tile_dim1 = BN
            g1[5] = (unsigned)K;                          // dim0_stride = K
            g1[6] = 0u;
            g1[7] = 0u;                                   // dim1_stride unused
            asm volatile("tensor_load_to_lds %0, %1"
                         :: "s"(g0), "s"(g1) : "memory");
        }

        // ---- A tile: cvt f32 -> bf16 (overlaps the TDM transfer) ----
        const float4 av = *(const float4*)(A + aBase + (size_t)(k0 + acol));
        As[arow][acol + 0] = (__bf16)av.x;
        As[arow][acol + 1] = (__bf16)av.y;
        As[arow][acol + 2] = (__bf16)av.z;
        As[arow][acol + 3] = (__bf16)av.w;
        if (k0 + KC < K)     // pull next A slab toward L2 (global_prefetch_b8)
            __builtin_prefetch(A + aBase + (size_t)(k0 + KC), 0, 1);

        if (wave == 0)
            __builtin_amdgcn_s_wait_tensorcnt(0);   // TDM done before release
        __syncthreads();

        // ---- fragments per CDNA5 WMMA VGPR layouts ----
        union { uint4 u[2]; v16bf v; } af;
        // A 16x32: lane<16 holds K {0..7,16..23}; lane>=16 holds K {8..15,24..31}
        const __bf16* ap = &As[wm * 16 + l][half ? 8 : 0];
        af.u[0] = *(const uint4*)ap;
        af.u[1] = *(const uint4*)(ap + 16);

#pragma unroll
        for (int sub = 0; sub < 4; ++sub) {
            // B 32x16: lane identifies N; lanes<16 K {0..7,16..23}, lanes>=16 K {8..15,24..31}
            union { uint4 u[2]; v16bf v; } bf;
            const __bf16* bp = &Bs[wn * 64 + sub * 16 + l][half * 8];
            bf.u[0] = *(const uint4*)bp;
            bf.u[1] = *(const uint4*)(bp + 16);
            acc[sub] = __builtin_amdgcn_wmma_f32_16x16x32_bf16(
                           false, af.v, false, bf.v, (short)0, acc[sub],
                           false, false);
        }
        __syncthreads();   // all wmma consumed LDS; safe for next TDM write
    }

    // ---- epilogue: C layout VGPR r -> M = r (+8 for lanes>=16), N = lane%16 ----
    const int mOut = mBlock + wm * 16 + half * 8;
#pragma unroll
    for (int sub = 0; sub < 4; ++sub) {
        const int n = nBlock + wn * 64 + sub * 16 + l;
        const float bv = bias ? bias[n] : 0.0f;
#pragma unroll
        for (int r = 0; r < 8; ++r)
            C[(size_t)(mOut + r) * (size_t)ldc + n] = acc[sub][r] + bv;
    }
}

// ---------------------------------------------------------------------------
// Prep / elementwise kernels
// ---------------------------------------------------------------------------
__global__ void k_convert(__bf16* dst, const float* src, int n) {
    int i = blockIdx.x * 256 + threadIdx.x;
    if (i < n) dst[i] = (__bf16)src[i];
}

// Wcat[n*XK + k] = k < 1536 ? W_ih[n,k] : W_hh[n,k-1536]   (bf16)
__global__ void k_wcat(__bf16* dst, const float* Wih, const float* Whh) {
    int i = blockIdx.x * 256 + threadIdx.x;
    if (i >= GN * XK) return;
    int n = i / XK, k = i % XK;
    float v = (k < EE + HEE) ? Wih[(size_t)n * (EE + HEE) + k]
                             : Whh[(size_t)n * HH + (k - EE - HEE)];
    dst[i] = (__bf16)v;
}

__global__ void k_bias_sum(float* dst, const float* a, const float* b, int n) {
    int i = blockIdx.x * 256 + threadIdx.x;
    if (i < n) dst[i] = a[i] + b[i];
}

__global__ void k_zero(float* p, int n) {
    int i = blockIdx.x * 256 + threadIdx.x;
    if (i < n) p[i] = 0.0f;
}

__global__ void k_copy(float* dst, const float* src, int n) {
    int i = blockIdx.x * 256 + threadIdx.x;
    if (i < n) dst[i] = src[i];
}

// xcat[b, 0:2560] = [ emb_table[inputs[t,b]] | cont[b] | h[b] ]
__global__ void k_concat_x(float* xcat, const int* inputs, const float* emb,
                           const float* cont, const float* h, int t) {
    int i = blockIdx.x * 256 + threadIdx.x;
    if (i >= BB * XK) return;
    int b = i / XK, k = i % XK;
    float v;
    if (k < EE)            v = emb[(size_t)inputs[t * BB + b] * EE + k];
    else if (k < EE + HEE) v = cont[b * HEE + (k - EE)];
    else                   v = h[b * HH + (k - EE - HEE)];
    xcat[i] = v;
}

// hc[b, 0:2048] = [ h[b] | cont[b] ]
__global__ void k_concat_hc(float* hc, const float* h, const float* cont) {
    int i = blockIdx.x * 256 + threadIdx.x;
    if (i >= BB * (HH + HEE)) return;
    int b = i / (HH + HEE), k = i % (HH + HEE);
    hc[i] = (k < HH) ? h[b * HH + k] : cont[b * HEE + (k - HH)];
}

// LSTM cell (gates already include both biases from GEMM epilogue)
__global__ void k_lstm(const float* gates, float* h, float* c, float* raw_t) {
    int i = blockIdx.x * 256 + threadIdx.x;
    if (i >= BB * HH) return;
    int b = i / HH, j = i % HH;
    const float* g = gates + (size_t)b * GN;
    float ig = 1.0f / (1.0f + __expf(-g[j]));
    float fg = 1.0f / (1.0f + __expf(-g[HH + j]));
    float gg = tanhf(g[2 * HH + j]);
    float og = 1.0f / (1.0f + __expf(-g[3 * HH + j]));
    float cv = fg * c[i] + ig * gg;
    float hv = og * tanhf(cv);
    c[i] = cv; h[i] = hv; raw_t[i] = hv;
}

// sc[b,s] = sum_d v[d] * tanh(q[b,d] + ctx_proj[b,s,d])  (+ -inf mask)
__global__ void k_scores(const float* q, const float* ctxproj, const float* v,
                         const unsigned char* mask, float* sc) {
    __shared__ float red[256];
    int s = blockIdx.x, b = blockIdx.y, tid = threadIdx.x;
    const float* cp = ctxproj + ((size_t)(b * SS + s)) * HH;
    const float* qb = q + (size_t)b * HH;
    float acc = 0.0f;
    for (int d = tid; d < HH; d += 256)
        acc += v[d] * tanhf(qb[d] + cp[d]);
    red[tid] = acc; __syncthreads();
    for (int off = 128; off; off >>= 1) {
        if (tid < off) red[tid] += red[tid + off];
        __syncthreads();
    }
    if (tid == 0)
        sc[b * SS + s] = mask[b * SS + s] ? -__builtin_inff() : red[0];
}

// row softmax over S=256 (one block per batch row)
__global__ void k_softmax(const float* sc, float* attn) {
    __shared__ float red[SS];
    int b = blockIdx.x, s = threadIdx.x;
    float x = sc[b * SS + s];
    red[s] = x; __syncthreads();
    for (int off = 128; off; off >>= 1) {
        if (s < off) red[s] = fmaxf(red[s], red[s + off]);
        __syncthreads();
    }
    float mx = red[0]; __syncthreads();
    float e = __expf(x - mx);
    red[s] = e; __syncthreads();
    for (int off = 128; off; off >>= 1) {
        if (s < off) red[s] += red[s + off];
        __syncthreads();
    }
    attn[b * SS + s] = e / red[0];
}

// cont[b,he] = sum_s attn[b,s] * context[s,b,he]
__global__ void k_context(const float* attn, const float* ctx,
                          float* cont, float* content_t) {
    __shared__ float a[SS];
    int b  = blockIdx.y;
    int he = blockIdx.x * 256 + threadIdx.x;
    a[threadIdx.x] = attn[b * SS + threadIdx.x];
    __syncthreads();
    float acc = 0.0f;
    for (int s = 0; s < SS; ++s)
        acc += a[s] * ctx[((size_t)s * BB + b) * HEE + he];
    cont[b * HEE + he]      = acc;
    content_t[b * HEE + he] = acc;
}

// ---------------------------------------------------------------------------
// Host orchestration (graph-capture safe: stream launches only)
// ---------------------------------------------------------------------------
extern "C" void kernel_launch(void* const* d_in, const int* in_sizes, int n_in,
                              void* d_out, int out_size, void* d_ws, size_t ws_size,
                              hipStream_t stream)
{
    const int*           inputs  = (const int*)d_in[0];
    const float*         context = (const float*)d_in[1];
    const unsigned char* cmask   = (const unsigned char*)d_in[2];
    const float*         emb     = (const float*)d_in[3];
    const float*         W_ih    = (const float*)d_in[4];
    const float*         W_hh    = (const float*)d_in[5];
    const float*         b_ih    = (const float*)d_in[6];
    const float*         b_hh    = (const float*)d_in[7];
    const float*         Wi      = (const float*)d_in[8];
    const float*         Wc      = (const float*)d_in[9];
    const float*         b_c     = (const float*)d_in[10];
    const float*         v       = (const float*)d_in[11];
    const float*         W_out   = (const float*)d_in[12];
    const float*         b_out   = (const float*)d_in[13];

    // workspace carve-up (256B aligned)
    char* ws = (char*)d_ws;
    size_t off = 0;
    auto alloc = [&](size_t bytes) -> char* {
        char* p = ws + off;
        off += (bytes + 255) & ~(size_t)255;
        return p;
    };
    float*  ctxproj = (float*) alloc((size_t)BB * SS * HH * 4);   // 64 MB
    __bf16* Wcat    = (__bf16*)alloc((size_t)GN * XK * 2);        // 20 MB
    __bf16* WiB     = (__bf16*)alloc((size_t)HH * HH * 2);
    __bf16* WcB     = (__bf16*)alloc((size_t)HH * HEE * 2);
    __bf16* WoB     = (__bf16*)alloc((size_t)HH * (HH + HEE) * 2);
    float*  biasg   = (float*) alloc((size_t)GN * 4);
    float*  xcat    = (float*) alloc((size_t)BB * XK * 4);
    float*  hc      = (float*) alloc((size_t)BB * (HH + HEE) * 4);
    float*  gates   = (float*) alloc((size_t)BB * GN * 4);
    float*  qb      = (float*) alloc((size_t)BB * HH * 4);
    float*  hbuf    = (float*) alloc((size_t)BB * HH * 4);
    float*  cbuf    = (float*) alloc((size_t)BB * HH * 4);
    float*  contbuf = (float*) alloc((size_t)BB * HEE * 4);
    float*  scbuf   = (float*) alloc((size_t)BB * SS * 4);
    (void)ws_size; (void)in_sizes; (void)n_in; (void)out_size;

    // output regions (reference tuple order, flattened)
    float* outp   = (float*)d_out;
    float* rawO   = outp;
    float* outO   = rawO   + (size_t)TT * BB * HH;
    float* scoreO = outO   + (size_t)TT * BB * HH;
    float* contO  = scoreO + (size_t)TT * BB * SS;
    float* hfO    = contO  + (size_t)TT * BB * HEE;
    float* cfO    = hfO    + (size_t)BB * HH;

    // ---- prep: bf16 weight packs, bias sum, zero states ----
    k_wcat<<<(GN * XK + 255) / 256, 256, 0, stream>>>(Wcat, W_ih, W_hh);
    k_convert<<<(HH * HH + 255) / 256, 256, 0, stream>>>(WiB, Wi, HH * HH);
    k_convert<<<(HH * HEE + 255) / 256, 256, 0, stream>>>(WcB, Wc, HH * HEE);
    k_convert<<<(HH * (HH + HEE) + 255) / 256, 256, 0, stream>>>(WoB, W_out, HH * (HH + HEE));
    k_bias_sum<<<(GN + 255) / 256, 256, 0, stream>>>(biasg, b_ih, b_hh, GN);
    k_zero<<<(BB * HH + 255) / 256, 256, 0, stream>>>(hbuf, BB * HH);
    k_zero<<<(BB * HH + 255) / 256, 256, 0, stream>>>(cbuf, BB * HH);
    k_zero<<<(BB * HEE + 255) / 256, 256, 0, stream>>>(contbuf, BB * HEE);

    // ---- hoisted ctx_proj = ctx @ Wc^T + b_c  (M=16384, N=1024, K=1024) ----
    {
        dim3 g(HH / BN, (BB * SS) / BM);  // (4, 512)
        // A row m = b*S+s maps to context[(s*B + b)*HE + k]
        k_gemm_bf16<<<g, 256, 0, stream>>>(ctxproj, HH, context,
                                           SS, (long long)BB * HEE, (long long)HEE,
                                           WcB, b_c, BB * SS, HH, HEE);
    }

    // ---- sequential decode scan ----
    for (int t = 0; t < TT; ++t) {
        k_concat_x<<<(BB * XK + 255) / 256, 256, 0, stream>>>(
            xcat, inputs, emb, contbuf, hbuf, t);

        dim3 gG(GN / BN, BB / BM);  // (16, 2) — gates: 64x4096x2560
        k_gemm_bf16<<<gG, 256, 0, stream>>>(gates, GN, xcat,
                                            1 << 30, (long long)XK, 0,
                                            Wcat, biasg, BB, GN, XK);

        k_lstm<<<(BB * HH + 255) / 256, 256, 0, stream>>>(
            gates, hbuf, cbuf, rawO + (size_t)t * BB * HH);

        dim3 gQ(HH / BN, BB / BM);  // (4, 2) — q = h @ Wi^T
        k_gemm_bf16<<<gQ, 256, 0, stream>>>(qb, HH, hbuf,
                                            1 << 30, (long long)HH, 0,
                                            WiB, nullptr, BB, HH, HH);

        dim3 gS(SS, BB);
        k_scores<<<gS, 256, 0, stream>>>(qb, ctxproj, v, cmask, scbuf);

        k_softmax<<<BB, SS, 0, stream>>>(scbuf, scoreO + (size_t)t * BB * SS);

        dim3 gC(HEE / 256, BB);
        k_context<<<gC, 256, 0, stream>>>(scoreO + (size_t)t * BB * SS, context,
                                          contbuf, contO + (size_t)t * BB * HEE);

        k_concat_hc<<<(BB * (HH + HEE) + 255) / 256, 256, 0, stream>>>(
            hc, hbuf, contbuf);

        // out = [h|cont] @ W_out^T + b_out  (64x1024x2048), straight to d_out
        k_gemm_bf16<<<gQ, 256, 0, stream>>>(outO + (size_t)t * BB * HH, HH, hc,
                                            1 << 30, (long long)(HH + HEE), 0,
                                            WoB, b_out, BB, HH, HH + HEE);
    }

    k_copy<<<(BB * HH + 255) / 256, 256, 0, stream>>>(hfO, hbuf, BB * HH);
    k_copy<<<(BB * HH + 255) / 256, 256, 0, stream>>>(cfO, cbuf, BB * HH);
}
